// GraphRulesMultiClass_65120294142437
// MI455X (gfx1250) — matrix operand order repaired and checked by reading, not compiled, not run on hardware
//
#include <hip/hip_runtime.h>
#include <hip/hip_bf16.h>

typedef __attribute__((ext_vector_type(16))) _Float16 v16h;
typedef __attribute__((ext_vector_type(8)))  float    v8f;

union U16h {
  uint4 q[2];
  v16h  v;
};

// ---------------------------------------------------------------------------
// WMMA fragment loaders per CDNA5 ISA 7.12.2 (wave32, 16x16x32 f16).
// A: 16x32 (MxK). lane<16: M=lane, needs K=[0..7] (V0..3) and K=[16..23] (V4..7)
//    lane>=16: M=lane-16, K=[8..15] and K=[24..31]
// -> two contiguous 16B runs per lane: base+row*ld+ksel and +16 halves further.
// ---------------------------------------------------------------------------
__device__ __forceinline__ v16h load_a_frag_vec(const _Float16* __restrict__ base,
                                                int ldHalves, int lane) {
  const _Float16* r = base + (size_t)(lane & 15) * ldHalves + ((lane >> 4) << 3);
  U16h u;
  u.q[0] = *reinterpret_cast<const uint4*>(r);
  u.q[1] = *reinterpret_cast<const uint4*>(r + 16);
  return u.v;
}

// B fragments pre-packed fragment-major: frag (kt,nt) holds 32 lanes x 16
// halves contiguous; lane's half j is K=32*kt+8*(lane>>4)+(j&7)+16*(j>>3) at column n0+(lane&15).
__device__ __forceinline__ v16h load_b_frag_packed(const _Float16* __restrict__ packed,
                                                   int frag, int lane) {
  const _Float16* r = packed + ((size_t)frag << 9) + ((size_t)lane << 4);
  U16h u;
  u.q[0] = *reinterpret_cast<const uint4*>(r);
  u.q[1] = *reinterpret_cast<const uint4*>(r + 8);
  return u.v;
}
// C/D: 16x16 f32, lane: N = lane&15, VGPR r: M = r + 8*(lane>>4)

// ---------------------------------------------------------------------------
// Small utility kernels
// ---------------------------------------------------------------------------
__global__ void k_fill(float* p, float v, long long n) {
  long long i = blockIdx.x * (long long)blockDim.x + threadIdx.x;
  if (i < n) p[i] = v;
}

__global__ void k_deg_count(const int* __restrict__ dst, float* __restrict__ deg, int E) {
  int e = blockIdx.x * blockDim.x + threadIdx.x;
  if (e < E) atomicAdd(&deg[dst[e]], 1.0f);
}

__global__ void k_rsqrt_inplace(float* p, int n) {
  int i = blockIdx.x * blockDim.x + threadIdx.x;
  if (i < n) p[i] = rsqrtf(p[i]);
}

// pack fp32 weight [K, F] into fragment-major f16 B layout.
// out[((kt*NT + nt)*32 + lane)*16 + j] = W[kt*32 + (lane>>4)*8 + (j&7) + (j>>3)*16][nt*16 + (lane&15)]
__global__ void k_pack_b(const float* __restrict__ w, _Float16* __restrict__ out,
                         int K, int Ktiles, int F) {
  const int NT = F >> 4;
  const int total = Ktiles * NT * 512;
  int idx = blockIdx.x * blockDim.x + threadIdx.x;
  if (idx >= total) return;
  const int j    = idx & 15;
  const int lane = (idx >> 4) & 31;
  const int rest = idx >> 9;
  const int nt   = rest % NT;
  const int kt   = rest / NT;
  const int k    = kt * 32 + ((lane >> 4) << 3) + (j & 7) + ((j >> 3) << 4);
  const int col  = nt * 16 + (lane & 15);
  out[idx] = (k < K) ? (_Float16)w[(size_t)k * F + col] : (_Float16)0.0f;
}

// pack Wl3 [64,3] fp32 -> 2 B frags (32x16 each) with cols >= 3 zeroed
__global__ void k_pack_b3(const float* __restrict__ w, _Float16* __restrict__ out) {
  int idx = blockIdx.x * blockDim.x + threadIdx.x;
  if (idx >= 1024) return;
  const int j    = idx & 15;
  const int lane = (idx >> 4) & 31;
  const int kt   = idx >> 9;
  const int k    = kt * 32 + ((lane >> 4) << 3) + (j & 7) + ((j >> 3) << 4);
  const int col  = lane & 15;
  out[idx] = (col < 3) ? (_Float16)w[(size_t)k * 3 + col] : (_Float16)0.0f;
}

// x [N,8] fp32 -> act [N,128] f16 with zero padding
__global__ void k_init_act(const float* __restrict__ x, _Float16* __restrict__ act, int N) {
  int idx = blockIdx.x * blockDim.x + threadIdx.x;
  if (idx >= N * 128) return;
  int n = idx >> 7;
  int j = idx & 127;
  act[idx] = (j < 8) ? (_Float16)x[(size_t)n * 8 + j] : (_Float16)0.0f;
}

// agg = h * dinv^2  (self-loop term; also initialization).  F = 1<<fshift.
__global__ void k_self_init(const float* __restrict__ h, const float* __restrict__ dinv,
                            float* __restrict__ agg, int N, int fshift) {
  long long idx = blockIdx.x * (long long)blockDim.x + threadIdx.x;
  if (idx >= ((long long)N << fshift)) return;
  int n = (int)(idx >> fshift);
  float di = dinv[n];
  agg[idx] = h[idx] * di * di;
}

// agg[dst] += h[src] * dinv[src]*dinv[dst], 4 features per thread
__global__ void k_scatter(const float* __restrict__ h, const int* __restrict__ src,
                          const int* __restrict__ dst, const float* __restrict__ dinv,
                          float* __restrict__ agg, int E, int fshift) {
  const int f4shift = fshift - 2;
  long long idx = blockIdx.x * (long long)blockDim.x + threadIdx.x;
  if (idx >= ((long long)E << f4shift)) return;
  int e = (int)(idx >> f4shift);
  int f = (int)(idx & ((1 << f4shift) - 1)) << 2;
  int s = src[e], d = dst[e];
  float nm = dinv[s] * dinv[d];
  const float* hs = h + ((size_t)s << fshift) + f;
  float* ad = agg + ((size_t)d << fshift) + f;
#pragma unroll
  for (int j = 0; j < 4; ++j) atomicAdd(&ad[j], hs[j] * nm);
}

// out16 = (f16) relu(agg + b)
__global__ void k_finalize(const float* __restrict__ agg, const float* __restrict__ b,
                           _Float16* __restrict__ out16, int N, int fshift) {
  long long idx = blockIdx.x * (long long)blockDim.x + threadIdx.x;
  if (idx >= ((long long)N << fshift)) return;
  int f = (int)(idx & ((1 << fshift) - 1));
  out16[idx] = (_Float16)fmaxf(agg[idx] + b[f], 0.0f);
}

// ---------------------------------------------------------------------------
// WMMA node GEMM: C[M,Fout] = A16[M, ld=128] @ Bpacked[KT x NTW*4 frags]
// 64 rows (4 M-tiles) per block; 4 waves split the N tiles (NTW tiles each);
// fully unrolled; all 4 A fragments issued before the 4 WMMAs per k-chunk so
// the scheduler can use graduated counter waits.
// A must have >= 64 rows of slop past M (allocated by host).
// ---------------------------------------------------------------------------
template <int KT, int NTW>
__global__ void __launch_bounds__(128)
k_gemm_wmma(const _Float16* __restrict__ A,
            const _Float16* __restrict__ Bp,
            float* __restrict__ C, int ldc, int M) {
  const int lane = threadIdx.x & 31;
  const int wave = threadIdx.x >> 5;
  const int m0   = blockIdx.x * 64;
  if (m0 >= M) return;
  constexpr int NTILES = NTW * 4;
#pragma unroll
  for (int i = 0; i < NTW; ++i) {
    const int nt = wave + i * 4;
    v8f acc[4];
    const v8f zero = {0.f, 0.f, 0.f, 0.f, 0.f, 0.f, 0.f, 0.f};
#pragma unroll
    for (int mt = 0; mt < 4; ++mt) acc[mt] = zero;
#pragma unroll
    for (int kc = 0; kc < KT; ++kc) {
      v16h b = load_b_frag_packed(Bp, kc * NTILES + nt, lane);
      v16h a[4];
#pragma unroll
      for (int mt = 0; mt < 4; ++mt)
        a[mt] = load_a_frag_vec(A + (size_t)(m0 + mt * 16) * 128 + kc * 32, 128, lane);
#pragma unroll
      for (int mt = 0; mt < 4; ++mt)
        acc[mt] = __builtin_amdgcn_wmma_f32_16x16x32_f16(false, a[mt], false, b,
                                                         (short)0, acc[mt], false, false);
    }
    const int col = nt * 16 + (lane & 15);
    const int rb  = (lane >> 4) * 8;
#pragma unroll
    for (int mt = 0; mt < 4; ++mt) {
      const int row0 = m0 + mt * 16;
      float* crow = C + (size_t)(row0 + rb) * ldc + col;
      if (row0 + 16 <= M) {
#pragma unroll
        for (int r = 0; r < 8; ++r) crow[(size_t)r * ldc] = acc[mt][r];
      } else {
#pragma unroll
        for (int r = 0; r < 8; ++r)
          if (row0 + rb + r < M) crow[(size_t)r * ldc] = acc[mt][r];
      }
    }
  }
}

// ---------------------------------------------------------------------------
// Fused edge MLP: per-wave 64-edge tiles (4 M-tiles), async global->LDS
// gather, 3 WMMA layers with fused bias+ReLU, WMMA 64->3 tail.
// ---------------------------------------------------------------------------
#define EMLP_WAVES 2
#define EMLP_ROWS  64                            // edges per wave tile
#define EMLP_TILE_H (EMLP_ROWS * 128)            // halves per wave tile
#define EMLP_SX_H   (EMLP_WAVES * EMLP_TILE_H)   // sX region size in halves

template <int MT, int NT, int KT>
__device__ __forceinline__ void mlp_layer_wmma(const _Float16* __restrict__ Xs,
                                               const _Float16* __restrict__ Wp,
                                               const float* __restrict__ bias,
                                               _Float16* __restrict__ Os,
                                               int lane) {
#pragma unroll
  for (int nt = 0; nt < NT; ++nt) {
    const float bv = bias[nt * 16 + (lane & 15)];
    const v8f binit = {bv, bv, bv, bv, bv, bv, bv, bv};
    v8f acc[MT];
#pragma unroll
    for (int mt = 0; mt < MT; ++mt) acc[mt] = binit;
#pragma unroll
    for (int kc = 0; kc < KT; ++kc) {
      v16h b = load_b_frag_packed(Wp, kc * NT + nt, lane);
      v16h a[MT];
#pragma unroll
      for (int mt = 0; mt < MT; ++mt)
        a[mt] = load_a_frag_vec(Xs + (size_t)(mt * 16) * 128 + kc * 32, 128, lane);
#pragma unroll
      for (int mt = 0; mt < MT; ++mt)
        acc[mt] = __builtin_amdgcn_wmma_f32_16x16x32_f16(false, a[mt], false, b,
                                                         (short)0, acc[mt], false, false);
    }
    const int col = nt * 16 + (lane & 15);
    const int rb  = (lane >> 4) * 8;
#pragma unroll
    for (int mt = 0; mt < MT; ++mt)
#pragma unroll
      for (int r = 0; r < 8; ++r)
        Os[(size_t)(mt * 16 + rb + r) * 128 + col] = (_Float16)fmaxf(acc[mt][r], 0.0f);
  }
}

__global__ void __launch_bounds__(EMLP_WAVES * 32)
k_edge_mlp(const _Float16* __restrict__ h16,      // [N,64]
           const int* __restrict__ src, const int* __restrict__ dst,
           const _Float16* __restrict__ Wp16,     // packed frags, 4x8
           const float* __restrict__ bp,
           const _Float16* __restrict__ Wl116,    // packed frags, 4x4
           const float* __restrict__ bl1,
           const _Float16* __restrict__ Wl216,    // packed frags, 2x4
           const float* __restrict__ bl2,
           const _Float16* __restrict__ Wl3p,     // packed frags, 2x1 (cols>=3 zero)
           const float* __restrict__ bl3,
           float* __restrict__ out,               // [E,3]
           int E) {
  // single flat LDS buffer so byte offsets for async loads are known (base 0)
  __shared__ __align__(16) _Float16 smem[2 * EMLP_SX_H];   // 64 KB
  _Float16* sX = smem;              // [EMLP_WAVES][64][128]
  _Float16* sT = smem + EMLP_SX_H;  // [EMLP_WAVES][64][128]

  const int lane = threadIdx.x & 31;
  const int wave = threadIdx.x >> 5;
  const int tile = blockIdx.x * EMLP_WAVES + wave;
  const int ntiles = (E + EMLP_ROWS - 1) / EMLP_ROWS;
  const bool active = tile < ntiles;
  const int e0 = tile * EMLP_ROWS;

  _Float16* sXw = sX + wave * EMLP_TILE_H;
  _Float16* sTw = sT + wave * EMLP_TILE_H;

  // --- async gather: concat(h[src],h[dst]) -> wave tile (64 edges x 128 f16)
  if (active) {
#pragma unroll
    for (int jb = 0; jb < 2 * EMLP_ROWS / 32; ++jb) {
      const int job  = jb * 32 + lane;    // 0 .. 2*EMLP_ROWS-1
      const int r    = job >> 1;          // edge row 0..63
      const int half = job & 1;           // 0 = src half, 1 = dst half
      int e = e0 + r;
      if (e >= E) e = E - 1;
      const int node = half ? dst[e] : src[e];
      const _Float16* gp = h16 + (size_t)node * 64;
      const unsigned base_elem = (unsigned)((wave * EMLP_ROWS + r) * 128 + half * 64);
#pragma unroll
      for (int q = 0; q < 8; ++q) {
        unsigned ldsoff = (base_elem + q * 8) * 2u;  // LDS byte address
        unsigned long long ga = (unsigned long long)(uintptr_t)(gp + q * 8);
        asm volatile("global_load_async_to_lds_b128 %0, %1, off"
                     :: "v"(ldsoff), "v"(ga)
                     : "memory");
      }
    }
  }
  asm volatile("s_wait_asynccnt 0x0" ::: "memory");
  __syncthreads();

  // layer P: [64x128] @ [128x128] + bp, relu  -> sT
  if (active) mlp_layer_wmma<4, 8, 4>(sXw, Wp16, bp, sTw, lane);
  __syncthreads();

  // layer L1: [64x128] @ [128x64] + bl1, relu -> sX (cols 0..63)
  if (active) mlp_layer_wmma<4, 4, 4>(sTw, Wl116, bl1, sXw, lane);
  __syncthreads();

  // layer L2: [64x64] @ [64x64] + bl2, relu   -> sT (cols 0..63)
  if (active) mlp_layer_wmma<4, 4, 2>(sXw, Wl216, bl2, sTw, lane);
  __syncthreads();

  // final 64 -> 3 via WMMA against zero-padded Wl3 (cols 3..15 = 0)
  if (active) {
    const int col = lane & 15;
    const int rb  = (lane >> 4) * 8;
    const float bv = (col < 3) ? bl3[col] : 0.0f;
    const v8f binit = {bv, bv, bv, bv, bv, bv, bv, bv};
    v8f acc[4];
#pragma unroll
    for (int mt = 0; mt < 4; ++mt) acc[mt] = binit;
#pragma unroll
    for (int kc = 0; kc < 2; ++kc) {
      v16h b = load_b_frag_packed(Wl3p, kc, lane);
      v16h a[4];
#pragma unroll
      for (int mt = 0; mt < 4; ++mt)
        a[mt] = load_a_frag_vec(sTw + (size_t)(mt * 16) * 128 + kc * 32, 128, lane);
#pragma unroll
      for (int mt = 0; mt < 4; ++mt)
        acc[mt] = __builtin_amdgcn_wmma_f32_16x16x32_f16(false, a[mt], false, b,
                                                         (short)0, acc[mt], false, false);
    }
    if (col < 3) {
#pragma unroll
      for (int mt = 0; mt < 4; ++mt)
#pragma unroll
        for (int r = 0; r < 8; ++r) {
          const int e = e0 + mt * 16 + rb + r;
          if (e < E) out[(size_t)e * 3 + col] = acc[mt][r];
        }
    }
  }
}

// ---------------------------------------------------------------------------
// host orchestration
// ---------------------------------------------------------------------------
extern "C" void kernel_launch(void* const* d_in, const int* in_sizes, int n_in,
                              void* d_out, int out_size, void* d_ws, size_t ws_size,
                              hipStream_t stream) {
  (void)n_in; (void)out_size; (void)ws_size;
  const float* x   = (const float*)d_in[0];
  const int*   ei  = (const int*)d_in[1];
  const int N = in_sizes[0] / 8;
  const int E = in_sizes[1] / 2;
  const int* src = ei;
  const int* dst = ei + E;
  const float* W1  = (const float*)d_in[2];  const float* b1  = (const float*)d_in[3];
  const float* W2  = (const float*)d_in[4];  const float* b2  = (const float*)d_in[5];
  const float* W3  = (const float*)d_in[6];  const float* b3  = (const float*)d_in[7];
  const float* Wp  = (const float*)d_in[8];  const float* bp  = (const float*)d_in[9];
  const float* Wl1 = (const float*)d_in[10]; const float* bl1 = (const float*)d_in[11];
  const float* Wl2 = (const float*)d_in[12]; const float* bl2 = (const float*)d_in[13];
  const float* Wl3 = (const float*)d_in[14]; const float* bl3 = (const float*)d_in[15];
  float* out = (float*)d_out;

  char* p = (char*)d_ws;
  auto take = [&](size_t bytes) -> char* {
    char* r = p;
    p += (bytes + 255) & ~(size_t)255;
    return r;
  };
  float*     dinv = (float*)take((size_t)N * 4);
  _Float16*  act  = (_Float16*)take((size_t)(N + 64) * 128 * 2);  // 64 slop rows
  float*     h    = (float*)take((size_t)N * 128 * 4);
  float*     agg  = (float*)take((size_t)N * 128 * 4);
  _Float16*  h16  = (_Float16*)take((size_t)N * 64 * 2);
  _Float16*  w1p  = (_Float16*)take((size_t)1 * 8 * 512 * 2);
  _Float16*  w2p  = (_Float16*)take((size_t)4 * 8 * 512 * 2);
  _Float16*  w3p  = (_Float16*)take((size_t)4 * 4 * 512 * 2);
  _Float16*  wpp  = (_Float16*)take((size_t)4 * 8 * 512 * 2);
  _Float16*  wl1p = (_Float16*)take((size_t)4 * 4 * 512 * 2);
  _Float16*  wl2p = (_Float16*)take((size_t)2 * 4 * 512 * 2);
  _Float16*  wl3p = (_Float16*)take((size_t)2 * 1 * 512 * 2);

  const int T = 256;

  // degree (with self loop) and D^{-1/2}
  k_fill<<<(N + T - 1) / T, T, 0, stream>>>(dinv, 1.0f, (long long)N);
  k_deg_count<<<(E + T - 1) / T, T, 0, stream>>>(dst, dinv, E);
  k_rsqrt_inplace<<<(N + T - 1) / T, T, 0, stream>>>(dinv, N);

  // weights -> packed f16 B fragments (K zero-padded to multiple of 32)
  k_pack_b<<<(1 * 8 * 512 + T - 1) / T, T, 0, stream>>>(W1, w1p, 8, 1, 128);
  k_pack_b<<<(4 * 8 * 512 + T - 1) / T, T, 0, stream>>>(W2, w2p, 128, 4, 128);
  k_pack_b<<<(4 * 4 * 512 + T - 1) / T, T, 0, stream>>>(W3, w3p, 128, 4, 64);
  k_pack_b<<<(4 * 8 * 512 + T - 1) / T, T, 0, stream>>>(Wp, wpp, 128, 4, 128);
  k_pack_b<<<(4 * 4 * 512 + T - 1) / T, T, 0, stream>>>(Wl1, wl1p, 128, 4, 64);
  k_pack_b<<<(2 * 4 * 512 + T - 1) / T, T, 0, stream>>>(Wl2, wl2p, 64, 2, 64);
  k_pack_b3<<<4, T, 0, stream>>>(Wl3, wl3p);

  // x -> padded f16 activations
  k_init_act<<<((long long)N * 128 + T - 1) / T, T, 0, stream>>>(x, act, N);

  const int Mblocks = (N + 63) / 64;
  auto conv_tail = [&](int Fout, int fshift, const float* bias, _Float16* o16) {
    long long nf = (long long)N << fshift;
    k_self_init<<<(unsigned)((nf + T - 1) / T), T, 0, stream>>>(h, dinv, agg, N, fshift);
    long long ef = (long long)E << (fshift - 2);
    k_scatter<<<(unsigned)((ef + T - 1) / T), T, 0, stream>>>(h, src, dst, dinv, agg, E, fshift);
    k_finalize<<<(unsigned)((nf + T - 1) / T), T, 0, stream>>>(agg, bias, o16, N, fshift);
  };
  // layer 1: K=8 (1 k-tile), Fout=128
  k_gemm_wmma<1, 2><<<Mblocks, 128, 0, stream>>>(act, w1p, h, 128, N);
  conv_tail(128, 7, b1, act);
  // layer 2: K=128 (4 k-tiles), Fout=128
  k_gemm_wmma<4, 2><<<Mblocks, 128, 0, stream>>>(act, w2p, h, 128, N);
  conv_tail(128, 7, b2, act);
  // layer 3: K=128 (4 k-tiles), Fout=64
  k_gemm_wmma<4, 1><<<Mblocks, 128, 0, stream>>>(act, w3p, h, 64, N);
  conv_tail(64, 6, b3, h16);

  // fused edge MLP
  const int etiles = (E + EMLP_ROWS - 1) / EMLP_ROWS;
  const int blocks = (etiles + EMLP_WAVES - 1) / EMLP_WAVES;
  k_edge_mlp<<<blocks, EMLP_WAVES * 32, 0, stream>>>(h16, src, dst, wpp, bp, wl1p, bl1,
                                                     wl2p, bl2, wl3p, bl3, out, E);
}
